// CrazyFusionModel_31714038514084
// MI455X (gfx1250) — matrix operand order repaired and checked by reading, not compiled, not run on hardware
//
#include <hip/hip_runtime.h>
#include <hip/hip_bf16.h>

// ---------------------------------------------------------------------------
// GNN (TransformerConv / GATConv x4 + LN/ReLU + mean-pool + MLP) for MI455X.
// Dense h@W GEMMs run on v_wmma_f32_16x16x32_f16 (wave32 WMMA, f32 accum).
// Edge softmax/aggregation are bandwidth-bound scalar kernels with atomics.
// ---------------------------------------------------------------------------

#define HID     128
#define IN_DIMC 92
#define OUT_DIMC 256
#define HEADS   8
#define DK      16
#define LDA     136   // padded f16 row pitch (136*2B = 272B -> 4-bank lane stride)

typedef __attribute__((ext_vector_type(16))) _Float16 v16h;
typedef __attribute__((ext_vector_type(8)))  _Float16 v8h;
typedef __attribute__((ext_vector_type(4)))  _Float16 v4h;
typedef __attribute__((ext_vector_type(8)))  float    v8f;

// order-preserving float <-> uint encoding for atomicMax on floats
__device__ __forceinline__ unsigned fenc(float f) {
  unsigned u = __float_as_uint(f);
  return (u & 0x80000000u) ? ~u : (u | 0x80000000u);
}
__device__ __forceinline__ float fdec(unsigned u) {
  return __uint_as_float((u & 0x80000000u) ? (u & 0x7fffffffu) : ~u);
}

// ---------------------------------------------------------------------------
// Y[N,128] = X[N,Kin] @ W[Kin,128] + bias   (Kin <= 128, zero-padded)
// Block = 256 threads = 8 waves; each wave owns a 16-wide column strip and
// keeps its 4 B-fragments (K-tiles of 32) resident in VGPRs for the whole
// kernel. A tiles (64 rows) staged f32->f16 in LDS per chunk.
// ---------------------------------------------------------------------------
__global__ __launch_bounds__(256) void gemm_x128_wmma(
    const float* __restrict__ X, const float* __restrict__ W,
    const float* __restrict__ bias, float* __restrict__ Y,
    int N, int Kin)
{
  __shared__ _Float16 Ws[128 * LDA];
  __shared__ _Float16 As[64 * LDA];

  const int tid   = threadIdx.x;
  const int lane  = tid & 31;
  const int wid   = tid >> 5;
  const int col   = lane & 15;
  const int khalf = lane >> 4;
  const int n0    = wid << 4;

  // stage weights as f16, zero-pad K rows beyond Kin
  for (int idx = tid; idx < 128 * 128; idx += 256) {
    int k = idx >> 7, n = idx & 127;
    float w = (k < Kin) ? W[k * 128 + n] : 0.0f;
    Ws[k * LDA + n] = (_Float16)w;
  }
  __syncthreads();

  // build resident B fragments (contiguous K order, same as A: lanes0-15 K=0..15,
  // lanes16-31 K=16..31, 2 K-rows packed per VGPR)
  v16h bf[4];
#pragma unroll
  for (int kt = 0; kt < 4; ++kt) {
    const int kb = kt * 32 + (khalf << 4);
#pragma unroll
    for (int p = 0; p < 8; ++p) {
      bf[kt][2 * p]     = Ws[(kb + 2 * p) * LDA + n0 + col];
      bf[kt][2 * p + 1] = Ws[(kb + 2 * p + 1) * LDA + n0 + col];
    }
  }
  const float bv = bias ? bias[n0 + col] : 0.0f;
  const int   m  = lane & 15;

  for (int m0 = blockIdx.x * 64; m0 < N; m0 += gridDim.x * 64) {
    const bool full = (m0 + 64 <= N);  // uniform per block
    __syncthreads();
    if (full && Kin == HID) {
      // vectorized staging: float4 global loads -> 4xf16 packed ds_store_b64
      const float4* X4 = (const float4*)X + (size_t)m0 * 32;
#pragma unroll
      for (int it = 0; it < 8; ++it) {
        int idx = tid + it * 256;          // 0..2047 float4s (64 rows x 32)
        float4 t = X4[idx];
        v4h h;
        h[0] = (_Float16)t.x; h[1] = (_Float16)t.y;
        h[2] = (_Float16)t.z; h[3] = (_Float16)t.w;
        int r = idx >> 5, c = (idx & 31) << 2;
        *(v4h*)(&As[r * LDA + c]) = h;
      }
    } else {
      for (int idx = tid; idx < 64 * 128; idx += 256) {
        int r = idx >> 7, c = idx & 127;
        int gr = m0 + r;
        float x = (gr < N && c < Kin) ? X[(size_t)gr * Kin + c] : 0.0f;
        As[r * LDA + c] = (_Float16)x;
      }
    }
    __syncthreads();

#pragma unroll
    for (int mt = 0; mt < 4; ++mt) {
      const int arow = (mt * 16 + m) * LDA;
      // preload all four A fragments into distinct registers so the eight
      // ds_load_b128 can pipeline ahead of the four chained WMMAs
      v16h a4[4];
#pragma unroll
      for (int kt = 0; kt < 4; ++kt) {
        // A layout: lane<16 -> K {0..15}; lane>=16 -> K {16..31}
        const v8h lo = *(const v8h*)(&As[arow + kt * 32 + khalf * 16]);
        const v8h hi = *(const v8h*)(&As[arow + kt * 32 + 8 + khalf * 16]);
#pragma unroll
        for (int j = 0; j < 8; ++j) { a4[kt][j] = lo[j]; a4[kt][8 + j] = hi[j]; }
      }
      v8f acc = {};
#pragma unroll
      for (int kt = 0; kt < 4; ++kt)
        acc = __builtin_amdgcn_wmma_f32_16x16x32_f16(
            false, a4[kt], false, bf[kt], (short)0, acc, false, false);

      // C layout: VGPR i -> row i + khalf*8, col = lane&15
      const int rb = m0 + mt * 16 + khalf * 8;
      if (full) {
        float* Yp = Y + (size_t)rb * HID + n0 + col;
#pragma unroll
        for (int i = 0; i < 8; ++i) Yp[i * HID] = acc[i] + bv;  // imm offsets
      } else {
#pragma unroll
        for (int i = 0; i < 8; ++i) {
          int row = rb + i;
          if (row < N) Y[(size_t)row * HID + n0 + col] = acc[i] + bv;
        }
      }
    }
  }
}

// ---------------------------------------------------------------------------
// fills
// ---------------------------------------------------------------------------
__global__ void fill_f32(float* p, float v, int n) {
  int i = blockIdx.x * blockDim.x + threadIdx.x;
  if (i < n) p[i] = v;
}
__global__ void fill_u32(unsigned* p, unsigned v, int n) {
  int i = blockIdx.x * blockDim.x + threadIdx.x;
  if (i < n) p[i] = v;
}

// ---------------------------------------------------------------------------
// TransformerConv logits: logit[e,h] = <q[dst], k[src]>_16 / sqrt(16)
// ---------------------------------------------------------------------------
__global__ void tc_logits_k(const float* __restrict__ Q, const float* __restrict__ K,
                            const int* __restrict__ src, const int* __restrict__ dst,
                            float* __restrict__ LOGI, unsigned* __restrict__ MAXB, int E)
{
  int idx = blockIdx.x * blockDim.x + threadIdx.x;
  if (idx >= E * HEADS) return;
  int e = idx >> 3, h = idx & 7;
  int s = src[e], d = dst[e];
  const float4* q = (const float4*)(Q + (size_t)d * HID + h * DK);
  const float4* k = (const float4*)(K + (size_t)s * HID + h * DK);
  float acc = 0.0f;
#pragma unroll
  for (int j = 0; j < 4; ++j) {
    float4 a = q[j], b = k[j];
    acc += a.x * b.x + a.y * b.y + a.z * b.z + a.w * b.w;
  }
  float lg = acc * 0.25f;  // 1/sqrt(DK)
  LOGI[idx] = lg;
  atomicMax(&MAXB[(size_t)d * HEADS + h], fenc(lg));
}

// ---------------------------------------------------------------------------
// GAT per-node head scores: a_s[n,h] = <xp[n,h,:], att_src[h]>, same for dst
// ---------------------------------------------------------------------------
__global__ void gat_ad_k(const float* __restrict__ XP,
                         const float* __restrict__ att_src, const float* __restrict__ att_dst,
                         float* __restrict__ AS, float* __restrict__ AD, int N)
{
  int idx = blockIdx.x * blockDim.x + threadIdx.x;
  if (idx >= N * HEADS) return;
  int n = idx >> 3, h = idx & 7;
  const float* xr = XP + (size_t)n * HID + h * DK;
  float s = 0.0f, d = 0.0f;
#pragma unroll
  for (int j = 0; j < DK; ++j) {
    float x = xr[j];
    s += x * att_src[h * DK + j];
    d += x * att_dst[h * DK + j];
  }
  AS[idx] = s;
  AD[idx] = d;
}

// ---------------------------------------------------------------------------
// GAT edge logits (with self-loops appended: e >= E -> node e-E)
// ---------------------------------------------------------------------------
__global__ void gat_logits_k(const float* __restrict__ AS, const float* __restrict__ AD,
                             const int* __restrict__ src, const int* __restrict__ dst,
                             float* __restrict__ LOGI, unsigned* __restrict__ MAXB,
                             int E, int TOT)
{
  int idx = blockIdx.x * blockDim.x + threadIdx.x;
  if (idx >= TOT * HEADS) return;
  int e = idx >> 3, h = idx & 7;
  int s = (e < E) ? src[e] : (e - E);
  int d = (e < E) ? dst[e] : (e - E);
  float r = AS[(size_t)s * HEADS + h] + AD[(size_t)d * HEADS + h];
  float v = (r > 0.0f) ? r : 0.2f * r;  // leaky_relu(0.2)
  LOGI[idx] = v;
  atomicMax(&MAXB[(size_t)d * HEADS + h], fenc(v));
}

// ---------------------------------------------------------------------------
// segment softmax pass 2: exp(logit - max) and per-(dst,head) sums
// ---------------------------------------------------------------------------
__global__ void expsum_k(float* __restrict__ LOGI, const unsigned* __restrict__ MAXB,
                         float* __restrict__ SUMB, const int* __restrict__ dst,
                         int E, int TOT)
{
  int idx = blockIdx.x * blockDim.x + threadIdx.x;
  if (idx >= TOT * HEADS) return;
  int e = idx >> 3, h = idx & 7;
  int d = (e < E) ? dst[e] : (e - E);
  float m = fdec(MAXB[(size_t)d * HEADS + h]);
  if (!(m > -1e30f && m < 1e30f)) m = 0.0f;  // no-edge segments: max -> 0
  float ex = __expf(LOGI[idx] - m);
  LOGI[idx] = ex;
  atomicAdd(&SUMB[(size_t)d * HEADS + h], ex);
}

// ---------------------------------------------------------------------------
// pass 3: agg[dst, h*16..] += alpha * V[src, h*16..]
// ---------------------------------------------------------------------------
__global__ void agg_k(const float* __restrict__ LOGI, const float* __restrict__ SUMB,
                      const float* __restrict__ V,
                      const int* __restrict__ src, const int* __restrict__ dst,
                      float* __restrict__ AGG, int E, int TOT)
{
  int idx = blockIdx.x * blockDim.x + threadIdx.x;
  if (idx >= TOT * HEADS) return;
  int e = idx >> 3, h = idx & 7;
  int s = (e < E) ? src[e] : (e - E);
  int d = (e < E) ? dst[e] : (e - E);
  float alpha = LOGI[idx] / (SUMB[(size_t)d * HEADS + h] + 1e-16f);
  const float4* vp = (const float4*)(V + (size_t)s * HID + h * DK);
  float* ap = AGG + (size_t)d * HID + h * DK;
#pragma unroll
  for (int j = 0; j < 4; ++j) {
    float4 t = vp[j];
    atomicAdd(&ap[4 * j + 0], alpha * t.x);
    atomicAdd(&ap[4 * j + 1], alpha * t.y);
    atomicAdd(&ap[4 * j + 2], alpha * t.z);
    atomicAdd(&ap[4 * j + 3], alpha * t.w);
  }
}

// ---------------------------------------------------------------------------
// out = relu(LayerNorm(agg + skip? + bias? + res) * g + b). Wave per node.
// ---------------------------------------------------------------------------
__global__ __launch_bounds__(256) void ln_relu_k(
    const float* __restrict__ agg, const float* __restrict__ skip,
    const float* __restrict__ bias, const float* __restrict__ res,
    const float* __restrict__ g, const float* __restrict__ bb,
    float* __restrict__ out, int N)
{
  int wid  = (blockIdx.x * blockDim.x + threadIdx.x) >> 5;
  int lane = threadIdx.x & 31;
  if (wid >= N) return;
  size_t base = (size_t)wid * HID;
  float v[4];
#pragma unroll
  for (int j = 0; j < 4; ++j) {
    int c = lane + 32 * j;
    float x = agg[base + c] + res[base + c];
    if (skip) x += skip[base + c];
    if (bias) x += bias[c];
    v[j] = x;
  }
  float s = v[0] + v[1] + v[2] + v[3];
#pragma unroll
  for (int off = 16; off; off >>= 1) s += __shfl_xor(s, off, 32);
  float mu = s * (1.0f / 128.0f);
  float q = 0.0f;
#pragma unroll
  for (int j = 0; j < 4; ++j) { float d = v[j] - mu; q += d * d; }
#pragma unroll
  for (int off = 16; off; off >>= 1) q += __shfl_xor(q, off, 32);
  float rs = rsqrtf(q * (1.0f / 128.0f) + 1e-5f);
#pragma unroll
  for (int j = 0; j < 4; ++j) {
    int c = lane + 32 * j;
    float y = (v[j] - mu) * rs * g[c] + bb[c];
    out[base + c] = fmaxf(y, 0.0f);
  }
}

// ---------------------------------------------------------------------------
// mean pool (sum part) + tiny MLP head
// ---------------------------------------------------------------------------
__global__ void pool_sum_k(const float* __restrict__ H, float* __restrict__ pooled, int N) {
  int c = threadIdx.x;  // 128 threads
  float s = 0.0f;
  for (int r = blockIdx.x; r < N; r += gridDim.x) s += H[(size_t)r * HID + c];
  atomicAdd(&pooled[c], s);
}

__global__ __launch_bounds__(256) void mlp_head_k(
    const float* __restrict__ pooled,
    const float* __restrict__ w1, const float* __restrict__ b1,
    const float* __restrict__ w2, const float* __restrict__ b2,
    float* __restrict__ out, float invN)
{
  __shared__ float pl[HID];
  __shared__ float z1[2 * HID];
  int t = threadIdx.x;
  if (t < HID) pl[t] = pooled[t] * invN;
  __syncthreads();
  float acc = b1[t];
  for (int k = 0; k < HID; ++k) acc += pl[k] * w1[k * 256 + t];
  z1[t] = fmaxf(acc, 0.0f);
  __syncthreads();
  float o = b2[t];
  for (int j = 0; j < 2 * HID; ++j) o += z1[j] * w2[j * 256 + t];
  out[t] = o;
}

// ---------------------------------------------------------------------------
// host orchestration
// ---------------------------------------------------------------------------
static inline int cdiv(int a, int b) { return (a + b - 1) / b; }

extern "C" void kernel_launch(void* const* d_in, const int* in_sizes, int n_in,
                              void* d_out, int out_size, void* d_ws, size_t ws_size,
                              hipStream_t stream) {
  (void)n_in; (void)out_size; (void)ws_size;
  const int N = in_sizes[0] / IN_DIMC;  // x: [N, 92]
  const int E = in_sizes[1] / 2;        // edge_index: [2, E] int32 (JAX x64 off)

  const float* X  = (const float*)d_in[0];
  const int*   EI = (const int*)d_in[1];
  const int* src = EI;
  const int* dst = EI + E;

  // params, insertion order of setup_inputs()
  const float* in_w = (const float*)d_in[2];
  const float* in_b = (const float*)d_in[3];
  // layer 0 (TC): wq,bq,wk,bk,wv,bv,wskip,bskip at 4..11
  // layer 1 (GAT): w,att_src,att_dst,bias at 12..15
  // layer 2 (TC): 16..23 ; layer 3 (GAT): 24..27
  const float* LNG[4] = {(const float*)d_in[28], (const float*)d_in[29],
                         (const float*)d_in[30], (const float*)d_in[31]};
  const float* LNB[4] = {(const float*)d_in[32], (const float*)d_in[33],
                         (const float*)d_in[34], (const float*)d_in[35]};
  const float* out_w1 = (const float*)d_in[36];
  const float* out_b1 = (const float*)d_in[37];
  const float* out_w2 = (const float*)d_in[38];
  const float* out_b2 = (const float*)d_in[39];

  // workspace layout
  size_t n128 = (size_t)N * HID;
  float* C0 = (float*)d_ws;
  float* C1 = C0 + n128;
  float* C2 = C1 + n128;
  float* C3 = C2 + n128;
  float* C4 = C3 + n128;
  float* C5 = C4 + n128;          // AGG
  float* AS = C5 + n128;          // N*8
  float* AD = AS + (size_t)N * HEADS;
  float* LOGI = AD + (size_t)N * HEADS;                 // (E+N)*8
  unsigned* MAXB = (unsigned*)(LOGI + (size_t)(E + N) * HEADS);  // N*8
  float* SUMB = (float*)(MAXB + (size_t)N * HEADS);     // N*8
  float* POOL = SUMB + (size_t)N * HEADS;               // 128

  const dim3 blk(256);
  const dim3 gGemm((unsigned)((cdiv(N, 64) < 320) ? cdiv(N, 64) : 320));
  const int nh  = N * HEADS;
  const int nhb = cdiv(nh, 256);
  const int nfb = cdiv((int)n128, 256);
  const dim3 gPost((unsigned)cdiv(N, 8));  // 8 waves/block, wave per node

  // input projection: C0 = x @ in_w + in_b
  gemm_x128_wmma<<<gGemm, blk, 0, stream>>>(X, in_w, in_b, C0, N, IN_DIMC);

  for (int layer = 0; layer < 4; ++layer) {
    const bool tc = (layer % 2 == 0);
    float* res = tc ? C0 : C1;   // even: C0 -> C1, odd: C1 -> C0
    float* nxt = tc ? C1 : C0;
    const int base = (layer == 0) ? 4 : (layer == 1) ? 12 : (layer == 2) ? 16 : 24;

    // per-layer segment-softmax state
    fill_u32<<<nhb, blk, 0, stream>>>(MAXB, 0x007fffffu, nh);  // enc(-inf)
    fill_f32<<<nhb, blk, 0, stream>>>(SUMB, 0.0f, nh);
    fill_f32<<<nfb, blk, 0, stream>>>(C5, 0.0f, (int)n128);

    if (tc) {
      const float* wq = (const float*)d_in[base + 0];
      const float* bq = (const float*)d_in[base + 1];
      const float* wk = (const float*)d_in[base + 2];
      const float* bk = (const float*)d_in[base + 3];
      const float* wv = (const float*)d_in[base + 4];
      const float* bv = (const float*)d_in[base + 5];
      const float* ws = (const float*)d_in[base + 6];
      const float* bs = (const float*)d_in[base + 7];
      // q=C1(=nxt, dead before post writes it), k=C2, v=C3, skip=C4
      gemm_x128_wmma<<<gGemm, blk, 0, stream>>>(res, wq, bq, C1, N, HID);
      gemm_x128_wmma<<<gGemm, blk, 0, stream>>>(res, wk, bk, C2, N, HID);
      gemm_x128_wmma<<<gGemm, blk, 0, stream>>>(res, wv, bv, C3, N, HID);
      gemm_x128_wmma<<<gGemm, blk, 0, stream>>>(res, ws, bs, C4, N, HID);

      const int eb = cdiv(E * HEADS, 256);
      tc_logits_k<<<eb, blk, 0, stream>>>(C1, C2, src, dst, LOGI, MAXB, E);
      expsum_k<<<eb, blk, 0, stream>>>(LOGI, MAXB, SUMB, dst, E, E);
      agg_k<<<eb, blk, 0, stream>>>(LOGI, SUMB, C3, src, dst, C5, E, E);

      ln_relu_k<<<gPost, blk, 0, stream>>>(C5, C4, nullptr, res,
                                           LNG[layer], LNB[layer], nxt, N);
    } else {
      const float* w    = (const float*)d_in[base + 0];
      const float* asrc = (const float*)d_in[base + 1];
      const float* adst = (const float*)d_in[base + 2];
      const float* bias = (const float*)d_in[base + 3];
      // xp = C2
      gemm_x128_wmma<<<gGemm, blk, 0, stream>>>(res, w, nullptr, C2, N, HID);
      gat_ad_k<<<nhb, blk, 0, stream>>>(C2, asrc, adst, AS, AD, N);

      const int TOT = E + N;  // edges + self-loops
      const int eb = cdiv(TOT * HEADS, 256);
      gat_logits_k<<<eb, blk, 0, stream>>>(AS, AD, src, dst, LOGI, MAXB, E, TOT);
      expsum_k<<<eb, blk, 0, stream>>>(LOGI, MAXB, SUMB, dst, E, TOT);
      agg_k<<<eb, blk, 0, stream>>>(LOGI, SUMB, C2, src, dst, C5, E, TOT);

      ln_relu_k<<<gPost, blk, 0, stream>>>(C5, nullptr, bias, res,
                                           LNG[layer], LNB[layer], nxt, N);
    }
  }

  // mean pool (final h lives in C0 after layer 3) + MLP head
  fill_f32<<<1, 128, 0, stream>>>(POOL, 0.0f, HID);
  pool_sum_k<<<256, 128, 0, stream>>>(C0, POOL, N);
  mlp_head_k<<<1, blk, 0, stream>>>(POOL, out_w1, out_b1, out_w2, out_b2,
                                    (float*)d_out, 1.0f / (float)N);
}
